// TrajectoryPredictor_66125316489732
// MI455X (gfx1250) — matrix operand order repaired and checked by reading, not compiled, not run on hardware
//
#include <hip/hip_runtime.h>

// ---------------------------------------------------------------------------
// Types for CDNA5 WMMA
// ---------------------------------------------------------------------------
typedef unsigned short u16;
typedef __attribute__((ext_vector_type(16))) __bf16 v16bf;
typedef __attribute__((ext_vector_type(8)))  float  v8f;
typedef __attribute__((ext_vector_type(4)))  unsigned int v4u;

#define AS1 __attribute__((address_space(1)))

union FragU { v4u u2[2]; v16bf v; };

#define WMMA_BF16(a, b, c) \
  __builtin_amdgcn_wmma_f32_16x16x32_bf16(false, (a), false, (b), (short)0, (c), false, false)

// Force-global casts (inttoptr round trip: always legal, yields global_load/store)
__device__ __forceinline__ const AS1 float* gcf(const float* p) {
  return (const AS1 float*)(unsigned long long)p;
}
__device__ __forceinline__ AS1 float* gf(float* p) {
  return (AS1 float*)(unsigned long long)p;
}
__device__ __forceinline__ const AS1 u16* gcu(const u16* p) {
  return (const AS1 u16*)(unsigned long long)p;
}
__device__ __forceinline__ AS1 u16* gu(u16* p) {
  return (AS1 u16*)(unsigned long long)p;
}

__device__ __forceinline__ float bf2f(u16 h) {
  return __uint_as_float(((unsigned)h) << 16);
}
__device__ __forceinline__ u16 f2bf(float f) {
  unsigned u = __float_as_uint(f);
  unsigned r = u + 0x7FFFu + ((u >> 16) & 1u);   // round-to-nearest-even
  return (u16)(r >> 16);
}
__device__ __forceinline__ float eluf(float x) {
  return x > 0.f ? x : (__expf(x) - 1.f);
}
__device__ __forceinline__ float sigm(float x) {
  return 1.f / (1.f + __expf(-x));
}

// A-fragment: 16x32 bf16 tile from LDS (row-major, stride lda in elements).
// Lane layout (matches pretiled B): lanes 0-15 row M=lane, K 0..15;
// lanes 16-31 row M=lane-16, K 16..31.
__device__ __forceinline__ v16bf load_a_frag(const u16* A, int lda, int row0,
                                             int k0, int lane) {
  const u16* p = A + (size_t)(row0 + (lane & 15)) * lda + k0 + ((lane >> 4) << 4);
  FragU f;
  f.u2[0] = *(const v4u*)p;
  f.u2[1] = *(const v4u*)(p + 8);
  return f.v;
}

// B-fragment: pre-tiled bf16 weights in global memory. Tile (kt,nt) stores
// 32 lanes x 16 contiguous bf16 values: lane l = column nt*16+(l&15),
// K = kt*32 + (l>>4)*16 + j  (j = 0..15). Two global_load_b128 per lane.
__device__ __forceinline__ v16bf load_b_frag(const u16* Wt, int ktiles, int kt,
                                             int nt, int lane) {
  const AS1 u16* p = gcu(Wt) + ((size_t)(nt * ktiles + kt) * 32 + lane) * 16;
  FragU f;
  f.u2[0] = *(const AS1 v4u*)p;
  f.u2[1] = *(const AS1 v4u*)(p + 8);
  return f.v;
}

// ---------------------------------------------------------------------------
// Generic LDS-A x pretiled-B GEMM, compile-time shape for full K unroll.
// Output 16x16 tiles distributed over 8 waves. Epilogue adds bias, applies
// optional ELU, stores bf16 and/or fp32.
// ---------------------------------------------------------------------------
template <int KT, int MT, int NT, int ACT>
__device__ __forceinline__ void gemm_tile(const u16* A, int lda, const u16* Wt,
                                          const float* bias, u16* dstB,
                                          float* dstF, int ldd, int wave,
                                          int lane) {
  constexpr int total = MT * NT;
  for (int t = wave; t < total; t += 8) {
    const int mt = t % MT;
    const int nt = t / MT;
    v8f acc = {};
#pragma unroll
    for (int kt = 0; kt < KT; ++kt) {
      v16bf a = load_a_frag(A, lda, mt * 16, kt * 32, lane);
      v16bf b = load_b_frag(Wt, KT, kt, nt, lane);
      acc = WMMA_BF16(a, b, acc);
    }
    const int col = nt * 16 + (lane & 15);
    const float bv = bias ? gcf(bias)[col] : 0.f;
    const int rbase = mt * 16 + ((lane >> 4) << 3);
#pragma unroll
    for (int i = 0; i < 8; ++i) {
      float v = acc[i] + bv;
      if (ACT == 1) v = eluf(v);
      const int row = rbase + i;
      if (dstB) dstB[(size_t)row * ldd + col] = f2bf(v);
      if (dstF) dstF[(size_t)row * ldd + col] = v;
    }
  }
}

// ---------------------------------------------------------------------------
// Fused GRU layer: M=32 rows, HID=256, gates 3x256. Each wave owns a set of
// (row-tile, gate-col-tile) pairs and accumulates all 6 gate tiles
// (gi_r, gi_z, gi_n, gh_r, gh_z, gh_n) so gate math stays in registers.
// ---------------------------------------------------------------------------
__device__ void gru_layer(const u16* X, const u16* H, const u16* Wih,
                          const u16* Whh, const float* bih, const float* bhh,
                          u16* Hn, int wave, int lane) {
  for (int g = wave; g < 32; g += 8) {
    const int mt = g & 1;       // row tile (0..1)
    const int ct = g >> 1;      // gate column tile (0..15) => cols ct*16..
    v8f ai0 = {}, ai1 = {}, ai2 = {}, ah0 = {}, ah1 = {}, ah2 = {};
#pragma unroll
    for (int kt = 0; kt < 8; ++kt) {
      v16bf ax = load_a_frag(X, 256, mt * 16, kt * 32, lane);
      v16bf ah = load_a_frag(H, 256, mt * 16, kt * 32, lane);
      v16bf b0 = load_b_frag(Wih, 8, kt, ct, lane);
      v16bf b1 = load_b_frag(Wih, 8, kt, 16 + ct, lane);
      v16bf b2 = load_b_frag(Wih, 8, kt, 32 + ct, lane);
      ai0 = WMMA_BF16(ax, b0, ai0);
      ai1 = WMMA_BF16(ax, b1, ai1);
      ai2 = WMMA_BF16(ax, b2, ai2);
      v16bf c0 = load_b_frag(Whh, 8, kt, ct, lane);
      v16bf c1 = load_b_frag(Whh, 8, kt, 16 + ct, lane);
      v16bf c2 = load_b_frag(Whh, 8, kt, 32 + ct, lane);
      ah0 = WMMA_BF16(ah, c0, ah0);
      ah1 = WMMA_BF16(ah, c1, ah1);
      ah2 = WMMA_BF16(ah, c2, ah2);
    }
    const int col = ct * 16 + (lane & 15);
    const int rbase = mt * 16 + ((lane >> 4) << 3);
    const AS1 float* gbih = gcf(bih);
    const AS1 float* gbhh = gcf(bhh);
    const float br = gbih[col], bz = gbih[col + 256], bn = gbih[col + 512];
    const float cr = gbhh[col], cz = gbhh[col + 256], cn = gbhh[col + 512];
#pragma unroll
    for (int i = 0; i < 8; ++i) {
      const int row = rbase + i;
      const float ir = ai0[i] + br, iz = ai1[i] + bz, in = ai2[i] + bn;
      const float hr = ah0[i] + cr, hz = ah1[i] + cz, hn = ah2[i] + cn;
      const float r = sigm(ir + hr);
      const float zg = sigm(iz + hz);
      const float n = tanhf(in + r * hn);
      const float hold = bf2f(H[(size_t)row * 256 + col]);
      Hn[(size_t)row * 256 + col] = f2bf((1.f - zg) * n + zg * hold);
    }
  }
}

// ---------------------------------------------------------------------------
// Prediction heads: feat(32x128) @ W(128x16-padded) -> global out (B,T,10).
// ---------------------------------------------------------------------------
__device__ void head_gemm(const u16* A, const u16* Wt, const float* bias,
                          float* out, int t, int clip, int wave, int lane) {
  for (int mt = wave; mt < 2; mt += 8) {
    v8f acc = {};
#pragma unroll
    for (int kt = 0; kt < 4; ++kt) {
      v16bf a = load_a_frag(A, 128, mt * 16, kt * 32, lane);
      v16bf b = load_b_frag(Wt, 4, kt, 0, lane);
      acc = WMMA_BF16(a, b, acc);
    }
    const int col = lane & 15;
    if (col < 10) {
      const float bv = gcf(bias)[col];
      const int rbase = mt * 16 + ((lane >> 4) << 3);
      AS1 float* gout = gf(out);
#pragma unroll
      for (int i = 0; i < 8; ++i) {
        float v = acc[i] + bv;
        if (clip) v = fminf(fmaxf(v, -5.f), 2.f);
        const int row = rbase + i;
        gout[((size_t)row * 20 + t) * 10 + col] = v;
      }
    }
  }
}

// ---------------------------------------------------------------------------
// Pre-tile fp32 weight (K x N, row-major) into WMMA bf16 B-fragment layout.
// dst element index = ((nt*ktiles+kt)*32 + lane)*16 + j  maps to
// src[kt*32 + (lane>>4)*16 + j][nt*16 + (lane&15)]   (zero-padded).
// ---------------------------------------------------------------------------
__global__ void pretile_kernel(const float* __restrict__ src, int K, int N,
                               u16* __restrict__ dst, int ktiles, int ntiles) {
  const int total = ktiles * ntiles * 512;
  const int idx = blockIdx.x * 256 + threadIdx.x;
  if (idx >= total) return;
  const int j = idx & 15;
  const int lane = (idx >> 4) & 31;
  const int tileid = idx >> 9;
  const int kt = tileid % ktiles;
  const int nt = tileid / ktiles;
  const int row = kt * 32 + ((lane >> 4) << 4) + j;
  const int col = nt * 16 + (lane & 15);
  float v = (row < K && col < N) ? gcf(src)[(size_t)row * N + col] : 0.f;
  gu(dst)[idx] = f2bf(v);
}

// ---------------------------------------------------------------------------
// Main kernel: one workgroup (8 wave32) owns 32 batch rows for all 20 steps.
// ---------------------------------------------------------------------------
#define SMEM_BYTES 195584

__global__ __launch_bounds__(256) void traj_vae_kernel(
    const float* __restrict__ past, const float* __restrict__ fut,
    const float* __restrict__ ctx, const float* __restrict__ eps,
    const float* enc_b1, const float* enc_b2, const float* q_b,
    const float* k_b, const float* v_b, const float* o_b,
    const float* ln_g, const float* ln_b,
    const float* g0_bih, const float* g0_bhh,
    const float* g1_bih, const float* g1_bhh,
    const float* post_b1, const float* post_b2,
    const float* pri_b1, const float* pri_b2,
    const float* dec_b1, const float* dec_b2,
    const float* mean_b, const float* lstd_b,
    const u16* W_enc1, const u16* W_enc2, const u16* W_q, const u16* W_k,
    const u16* W_v, const u16* W_o, const u16* W_g0i, const u16* W_g0h,
    const u16* W_g1i, const u16* W_g1h, const u16* W_post1, const u16* W_post2,
    const u16* W_pri1, const u16* W_pri2, const u16* W_dec1, const u16* W_dec2,
    const u16* W_mean, const u16* W_lstd,
    float* __restrict__ out_pm, float* __restrict__ out_pl,
    float* __restrict__ out_kl) {
  extern __shared__ char smem[];
  const int tid = threadIdx.x;
  const int wave = tid >> 5;
  const int lane = tid & 31;
  const int b0 = blockIdx.x * 32;

  // Persistent LDS regions
  u16* sIn    = (u16*)(smem + 0);        // 32x32
  u16* emb1   = (u16*)(smem + 2048);     // 32x128
  u16* emb    = (u16*)(smem + 10240);    // 32x128
  u16* qB     = (u16*)(smem + 18432);    // 32x64
  u16* interB = (u16*)(smem + 22528);    // 32x64
  u16* xB     = (u16*)(smem + 26624);    // 32x256
  u16* h0B    = (u16*)(smem + 43008);    // 32x256
  u16* h1B    = (u16*)(smem + 59392);    // 32x256
  u16* hN     = (u16*)(smem + 75776);    // 32x256
  u16* zB     = (u16*)(smem + 92160);    // 32x64
  char* ovr = smem + 96256;
  // Overlay A: attention scratch
  u16*   ctxB    = (u16*)(ovr + 0);      // 256x32 (K padded)
  u16*   kB      = (u16*)(ovr + 16384);  // 256x64
  u16*   vB      = (u16*)(ovr + 49152);  // 256x64
  float* maskS   = (float*)(ovr + 81920);
  float* scoresS = (float*)(ovr + 82944);
  float* attnS   = (float*)(ovr + 84992);
  u16*   oB      = (u16*)(ovr + 87040);  // 32x64
  float* interF  = (float*)(ovr + 91136);// 32x64
  // Overlay B: posterior/prior/decoder scratch (disjoint lifetime)
  u16*   catB   = (u16*)(ovr + 0);       // 32x384
  u16*   hid256 = (u16*)(ovr + 24576);   // 32x256
  float* postF  = (float*)(ovr + 40960); // 32x128
  float* priF   = (float*)(ovr + 57344); // 32x128
  u16*   featB  = (u16*)(ovr + 73728);   // 32x128

  // init recurrent state
  for (int i = tid; i < 32 * 256; i += 256) { h0B[i] = 0; h1B[i] = 0; }
  for (int i = tid; i < 32 * 64; i += 256) zB[i] = 0;
  __syncthreads();

  for (int t = 0; t < 20; ++t) {
    // ---- stage state (pad 10 -> 32) ----
    for (int idx = tid; idx < 32 * 32; idx += 256) {
      const int r = idx >> 5, c = idx & 31;
      float v = 0.f;
      if (c < 10) {
        const int b = b0 + r;
        v = (t < 8) ? gcf(past)[((size_t)b * 8 + t) * 10 + c]
                    : gcf(fut)[((size_t)b * 12 + (t - 8)) * 10 + c];
      }
      sIn[idx] = f2bf(v);
    }
    __syncthreads();
    // ---- encoder MLP ----
    gemm_tile<1, 2, 8, 1>(sIn, 32, W_enc1, enc_b1, emb1, nullptr, 128, wave, lane);
    __syncthreads();
    gemm_tile<4, 2, 8, 1>(emb1, 128, W_enc2, enc_b2, emb, nullptr, 128, wave, lane);
    __syncthreads();
    // ---- q projection ----
    gemm_tile<4, 2, 4, 0>(emb, 128, W_q, q_b, qB, nullptr, 64, wave, lane);
    // ---- stage contexts + mask ----
    {
      const int idx = tid;  // exactly 256 = 32 rows * 8 neighbors
      const int r = idx >> 3, n = idx & 7;
      const AS1 float* cp =
          gcf(ctx) + (((size_t)(b0 + r) * 20 + t) * 8 + n) * 16;
      float ns = 0.f;
      for (int c = 0; c < 16; ++c) {
        float v = cp[c];
        ns += v * v;
        ctxB[(size_t)idx * 32 + c] = f2bf(v);
      }
      for (int c = 16; c < 32; ++c) ctxB[(size_t)idx * 32 + c] = 0;
      maskS[idx] = ns;
    }
    __syncthreads();
    // ---- K/V projections: M=256 rows ----
    gemm_tile<1, 16, 4, 0>(ctxB, 32, W_k, k_b, kB, nullptr, 64, wave, lane);
    gemm_tile<1, 16, 4, 0>(ctxB, 32, W_v, v_b, vB, nullptr, 64, wave, lane);
    __syncthreads();
    // ---- attention scores ----
    for (int idx = tid; idx < 512; idx += 256) {
      const int r = idx >> 4, hn2 = idx & 15;
      const int h = hn2 >> 3, n = hn2 & 7;
      float s = 0.f;
      for (int d = 0; d < 32; ++d)
        s += bf2f(qB[r * 64 + h * 32 + d]) * bf2f(kB[(r * 8 + n) * 64 + h * 32 + d]);
      s *= 0.17677669529663687f;  // 1/sqrt(32)
      if (maskS[r * 8 + n] < 1e-12f) s = -1e9f;
      scoresS[r * 16 + h * 8 + n] = s;
    }
    __syncthreads();
    if (tid < 64) {  // softmax over 8 neighbors per (row, head)
      const int r = tid >> 1, h = tid & 1;
      float* sc = scoresS + r * 16 + h * 8;
      float m = sc[0];
      for (int n = 1; n < 8; ++n) m = fmaxf(m, sc[n]);
      float ex[8], sum = 0.f;
      for (int n = 0; n < 8; ++n) { ex[n] = __expf(sc[n] - m); sum += ex[n]; }
      const float inv = 1.f / sum;
      for (int n = 0; n < 8; ++n) attnS[r * 16 + h * 8 + n] = ex[n] * inv;
    }
    __syncthreads();
    for (int idx = tid; idx < 32 * 64; idx += 256) {  // o = attn @ v
      const int r = idx >> 6, d = idx & 63, h = d >> 5;
      float acc = 0.f;
      for (int n = 0; n < 8; ++n)
        acc += attnS[r * 16 + h * 8 + n] * bf2f(vB[(r * 8 + n) * 64 + d]);
      oB[idx] = f2bf(acc);
    }
    __syncthreads();
    gemm_tile<2, 2, 4, 0>(oB, 64, W_o, o_b, nullptr, interF, 64, wave, lane);
    __syncthreads();
    if (tid < 32) {  // layernorm
      const float* x = interF + tid * 64;
      float mu = 0.f;
      for (int c = 0; c < 64; ++c) mu += x[c];
      mu *= (1.f / 64.f);
      float var = 0.f;
      for (int c = 0; c < 64; ++c) { float d = x[c] - mu; var += d * d; }
      var *= (1.f / 64.f);
      const float inv = rsqrtf(var + 1e-5f);
      for (int c = 0; c < 64; ++c)
        interB[tid * 64 + c] =
            f2bf(gcf(ln_g)[c] * (x[c] - mu) * inv + gcf(ln_b)[c]);
    }
    __syncthreads();
    // ---- GRU input x = [emb | inter | z_prev] ----
    for (int idx = tid; idx < 32 * 256; idx += 256) {
      const int r = idx >> 8, c = idx & 255;
      u16 v;
      if (c < 128)      v = emb[r * 128 + c];
      else if (c < 192) v = interB[r * 64 + (c - 128)];
      else              v = zB[r * 64 + (c - 192)];
      xB[idx] = v;
    }
    __syncthreads();
    gru_layer(xB, h0B, W_g0i, W_g0h, g0_bih, g0_bhh, hN, wave, lane);
    __syncthreads();
    for (int i = tid; i < 32 * 256; i += 256) h0B[i] = hN[i];
    __syncthreads();
    gru_layer(h0B, h1B, W_g1i, W_g1h, g1_bih, g1_bhh, hN, wave, lane);
    __syncthreads();
    for (int i = tid; i < 32 * 256; i += 256) h1B[i] = hN[i];
    __syncthreads();
    // ---- posterior: cat[ht, emb] ----
    for (int idx = tid; idx < 32 * 384; idx += 256) {
      const int r = idx / 384, c = idx % 384;
      catB[idx] = (c < 256) ? h1B[r * 256 + c] : emb[r * 128 + (c - 256)];
    }
    __syncthreads();
    gemm_tile<12, 2, 16, 1>(catB, 384, W_post1, post_b1, hid256, nullptr, 256, wave, lane);
    __syncthreads();
    gemm_tile<8, 2, 8, 0>(hid256, 256, W_post2, post_b2, nullptr, postF, 128, wave, lane);
    __syncthreads();
    // ---- prior ----
    gemm_tile<8, 2, 16, 1>(h1B, 256, W_pri1, pri_b1, hid256, nullptr, 256, wave, lane);
    __syncthreads();
    gemm_tile<8, 2, 8, 0>(hid256, 256, W_pri2, pri_b2, nullptr, priF, 128, wave, lane);
    __syncthreads();
    // ---- KL + reparameterized sample ----
    if (tid < 32) {
      const int b = b0 + tid;
      float kl = 0.f;
      for (int d = 0; d < 64; ++d) {
        const float qm = postF[tid * 128 + d];
        const float qls = fminf(fmaxf(postF[tid * 128 + 64 + d], -5.f), 2.f);
        const float pm2 = priF[tid * 128 + d];
        const float pls = fminf(fmaxf(priF[tid * 128 + 64 + d], -5.f), 2.f);
        const float v1 = __expf(2.f * qls), v2 = __expf(2.f * pls);
        const float dm = pm2 - qm;
        kl += 0.5f * (v1 / v2 + dm * dm / v2 - 1.f + 2.f * (pls - qls));
        const float zv =
            qm + __expf(qls) * gcf(eps)[((size_t)b * 20 + t) * 64 + d];
        const u16 zb = f2bf(zv);
        zB[tid * 64 + d] = zb;
        catB[tid * 384 + 256 + d] = zb;  // dec input cols 256..319 (ht kept)
      }
      gf(out_kl)[(size_t)b * 20 + t] = kl;
    }
    __syncthreads();
    // ---- decoder (catB cols 0..319 = [ht | z_new]) ----
    gemm_tile<10, 2, 16, 1>(catB, 384, W_dec1, dec_b1, hid256, nullptr, 256, wave, lane);
    __syncthreads();
    gemm_tile<8, 2, 8, 1>(hid256, 256, W_dec2, dec_b2, featB, nullptr, 128, wave, lane);
    __syncthreads();
    // ---- heads, write global ----
    head_gemm(featB, W_mean, mean_b, out_pm + (size_t)b0 * 200, t, 0, wave, lane);
    head_gemm(featB, W_lstd, lstd_b, out_pl + (size_t)b0 * 200, t, 1, wave, lane);
    __syncthreads();
  }
}

// ---------------------------------------------------------------------------
// Host entry
// ---------------------------------------------------------------------------
struct WSpec { int in_idx; int K; int N; };
static const WSpec g_wspecs[18] = {
    {4, 10, 128},  {6, 128, 128}, {8, 128, 64},  {10, 16, 64},  {12, 16, 64},
    {14, 64, 64},  {18, 256, 768},{19, 256, 768},{22, 256, 768},{23, 256, 768},
    {26, 384, 256},{28, 256, 128},{30, 256, 256},{32, 256, 128},{34, 320, 256},
    {36, 256, 128},{38, 128, 10}, {40, 128, 10}};

extern "C" void kernel_launch(void* const* d_in, const int* in_sizes, int n_in,
                              void* d_out, int out_size, void* d_ws,
                              size_t ws_size, hipStream_t stream) {
  (void)in_sizes; (void)n_in; (void)out_size; (void)ws_size;
  const float* past = (const float*)d_in[0];
  const float* fut  = (const float*)d_in[1];
  const float* ctx  = (const float*)d_in[2];
  const float* eps  = (const float*)d_in[3];

  // 1) pre-tile weights fp32 -> bf16 WMMA B-fragment layout in workspace
  u16* ws = (u16*)d_ws;
  const u16* Wp[18];
  size_t off = 0;
  for (int i = 0; i < 18; ++i) {
    const int K = g_wspecs[i].K, N = g_wspecs[i].N;
    const int kt = (K + 31) / 32, nt = (N + 15) / 16;
    const int total = kt * nt * 512;
    pretile_kernel<<<(total + 255) / 256, 256, 0, stream>>>(
        (const float*)d_in[g_wspecs[i].in_idx], K, N, ws + off, kt, nt);
    Wp[i] = ws + off;
    off += (size_t)total;
  }

  float* out = (float*)d_out;
  float* out_pm = out;
  float* out_pl = out + (size_t)8192 * 20 * 10;
  float* out_kl = out + (size_t)2 * 8192 * 20 * 10;

  (void)hipFuncSetAttribute((const void*)traj_vae_kernel,
                            hipFuncAttributeMaxDynamicSharedMemorySize,
                            SMEM_BYTES);

  // 2) main recurrent kernel: 256 workgroups x 32 batch rows
  traj_vae_kernel<<<256, 256, SMEM_BYTES, stream>>>(
      past, fut, ctx, eps,
      (const float*)d_in[5],  (const float*)d_in[7],  (const float*)d_in[9],
      (const float*)d_in[11], (const float*)d_in[13], (const float*)d_in[15],
      (const float*)d_in[16], (const float*)d_in[17],
      (const float*)d_in[20], (const float*)d_in[21],
      (const float*)d_in[24], (const float*)d_in[25],
      (const float*)d_in[27], (const float*)d_in[29],
      (const float*)d_in[31], (const float*)d_in[33],
      (const float*)d_in[35], (const float*)d_in[37],
      (const float*)d_in[39], (const float*)d_in[41],
      Wp[0], Wp[1], Wp[2], Wp[3], Wp[4], Wp[5], Wp[6], Wp[7], Wp[8], Wp[9],
      Wp[10], Wp[11], Wp[12], Wp[13], Wp[14], Wp[15], Wp[16], Wp[17],
      out_pm, out_pl, out_kl);
}
